// LLaMa_Block_SparseMoE_XL_46523085750486
// MI455X (gfx1250) — matrix operand order repaired and checked by reading, not compiled, not run on hardware
//
#include <hip/hip_runtime.h>
#include <hip/hip_bf16.h>
#include <stdint.h>

// Problem constants (from reference)
#define B_   2
#define T_   1024
#define D_   1024
#define H_   16
#define HD_  64
#define E_   8
#define FH_  4096
#define S_   (2 * T_)        // XL memory (T) + current (T)
#define M_   (B_ * T_)       // 2048 tokens

typedef __bf16 bf16_t;
typedef __attribute__((ext_vector_type(16))) __bf16 v16bf;
typedef __attribute__((ext_vector_type(8)))  float  v8f;

// ---------- CDNA5 async global->LDS copy (ASYNCcnt-tracked) ----------
#if defined(__has_builtin)
#  if __has_builtin(__builtin_amdgcn_global_load_async_to_lds_b128) && \
      __has_builtin(__builtin_amdgcn_s_wait_asynccnt)
#    define USE_ASYNC_LDS 1
#  endif
#endif

#ifdef USE_ASYNC_LDS
// Exact parameter type from the compiler diagnostic:
//   __attribute__((__vector_size__(4 * sizeof(int)))) int __device__ *
typedef int async_v4i __attribute__((vector_size(4 * sizeof(int))));
typedef __attribute__((address_space(1))) async_v4i* async_gptr;
typedef __attribute__((address_space(3))) async_v4i* async_lptr;
#endif

__device__ __forceinline__ void cp_async_b128(void* lds_dst, const void* gsrc) {
#ifdef USE_ASYNC_LDS
  __builtin_amdgcn_global_load_async_to_lds_b128(
      (async_gptr)gsrc, (async_lptr)lds_dst, 0, 0);
#else
  *(uint4*)lds_dst = *(const uint4*)gsrc;
#endif
}
__device__ __forceinline__ void cp_async_wait() {
#ifdef USE_ASYNC_LDS
  __builtin_amdgcn_s_wait_asynccnt(0);
#endif
}

// ---------- helpers ----------

__device__ __forceinline__ bf16_t f2bf(float f) {
  union { float f; uint32_t u; } a; a.f = f;
  uint32_t u = a.u;
  uint32_t lsb = (u >> 16) & 1u;
  u += 0x7fffu + lsb;                  // round-to-nearest-even
  union { uint16_t s; bf16_t b; } o; o.s = (uint16_t)(u >> 16);
  return o.b;
}

// A-fragment: 16(M) x 32(K) bf16 tile, row-major with element stride `stride`.
// ISA: lanes 0-15/16-31 hold M=0-15; VGPR0-3: K = hi*8+(0..7), VGPR4-7: K=16+hi*8+(0..7).
__device__ __forceinline__ v16bf load_a_frag(const bf16_t* base, int stride) {
  const int lane = threadIdx.x & 31;
  const int m  = lane & 15;
  const int hi = lane >> 4;
  union { v16bf v; uint32_t u[8]; } r;
  const bf16_t* row = base + m * stride;
#pragma unroll
  for (int g = 0; g < 2; ++g) {
    const int kb = g * 16 + hi * 8;
#pragma unroll
    for (int p = 0; p < 4; ++p)
      r.u[g * 4 + p] = *(const uint32_t*)(row + kb + 2 * p);
  }
  return r.v;
}

// B-fragment: 32(K) x 16(N) bf16, stored transposed in LDS as [n][k].
// Element i of lane L: k = (i>>3)*16 + 8*(L>>4) + (i&7), n = L&15.
__device__ __forceinline__ v16bf load_b_frag(const bf16_t* base, int stride) {
  const int lane = threadIdx.x & 31;
  const int n  = lane & 15;
  const int k0 = (lane >> 4) * 8;
  union { v16bf v; uint32_t u[8]; } r;
  const bf16_t* row = base + n * stride + k0;
#pragma unroll
  for (int p = 0; p < 8; ++p)
    r.u[p] = *(const uint32_t*)(row + (p >> 2) * 16 + 2 * (p & 3));
  return r.v;
}

__device__ __forceinline__ v8f wmma_bf16(v16bf a, v16bf b, v8f c) {
  return __builtin_amdgcn_wmma_f32_16x16x32_bf16(
      false, a, false, b, (short)0, c, false, false);
}

// ---------- elementwise kernels ----------

__global__ void cvt_f32_bf16_kernel(const float* __restrict__ in,
                                    bf16_t* __restrict__ out, int n) {
  for (int i = blockIdx.x * blockDim.x + threadIdx.x; i < n;
       i += gridDim.x * blockDim.x)
    out[i] = f2bf(in[i]);
}

__global__ void rmsnorm_kernel(const float* __restrict__ x,
                               const float* __restrict__ w,
                               float* __restrict__ of, bf16_t* __restrict__ ob) {
  const int row = blockIdx.x;
  const float* xr = x + (size_t)row * D_;
  __shared__ float red[256];
  float s = 0.f;
  for (int d = threadIdx.x; d < D_; d += 256) { float v = xr[d]; s += v * v; }
  red[threadIdx.x] = s; __syncthreads();
  for (int o = 128; o > 0; o >>= 1) {
    if (threadIdx.x < o) red[threadIdx.x] += red[threadIdx.x + o];
    __syncthreads();
  }
  const float inv = rsqrtf(red[0] / (float)D_ + 1e-6f);
  for (int d = threadIdx.x; d < D_; d += 256) {
    float v = xr[d] * inv * w[d];
    of[(size_t)row * D_ + d] = v;
    ob[(size_t)row * D_ + d] = f2bf(v);
  }
}

__global__ void final_add_kernel(const float* __restrict__ h,
                                 const float* __restrict__ m0p,
                                 const float* __restrict__ m1p,
                                 float* __restrict__ out, int n) {
  for (int i = blockIdx.x * blockDim.x + threadIdx.x; i < n;
       i += gridDim.x * blockDim.x)
    out[i] = h[i] + m0p[i] + m1p[i];
}

// ---------- generic GEMM ----------
// C[M,N](f32, +opt residual) = A[M,K](bf16) @ B[K,N](bf16)
// 256 threads = 8 waves; block tile 64(M) x 128(N); per-wave 32x32 (2x2 WMMA).
// Requires M % 64 == 0, N % 128 == 0, K % 32 == 0.
__global__ void gemm_bf16_kernel(const bf16_t* __restrict__ A,
                                 const bf16_t* __restrict__ Bm,
                                 const float* __restrict__ residual,
                                 float* __restrict__ C, int M, int N, int K) {
  __shared__ __align__(16) bf16_t sA[64][40];
  __shared__ __align__(16) bf16_t sB[128][40];   // transposed [n][k]
  const int tid  = threadIdx.x;
  const int wave = tid >> 5;
  const int lane = tid & 31;
  const int m0 = blockIdx.y * 64;
  const int n0 = blockIdx.x * 128;
  const int wm = (wave >> 2) * 32;
  const int wn = (wave & 3) * 32;
  const int am = tid >> 2, ac = (tid & 3) * 8;    // A staging: 1x b128 / thread
  v8f c00 = {}, c01 = {}, c10 = {}, c11 = {};
  for (int k0 = 0; k0 < K; k0 += 32) {
    cp_async_b128(&sA[am][ac], A + (size_t)(m0 + am) * K + k0 + ac);
    for (int i = tid; i < 128 * 32; i += 256) {
      int n = i & 127, k = i >> 7;
      sB[n][k] = Bm[(size_t)(k0 + k) * N + n0 + n];
    }
    cp_async_wait();
    __syncthreads();
    v16bf a0 = load_a_frag(&sA[wm][0], 40);
    v16bf a1 = load_a_frag(&sA[wm + 16][0], 40);
    v16bf b0 = load_b_frag(&sB[wn][0], 40);
    v16bf b1 = load_b_frag(&sB[wn + 16][0], 40);
    c00 = wmma_bf16(a0, b0, c00);
    c01 = wmma_bf16(a0, b1, c01);
    c10 = wmma_bf16(a1, b0, c10);
    c11 = wmma_bf16(a1, b1, c11);
    __syncthreads();
  }
  const int hi = lane >> 4, nl = lane & 15;
#pragma unroll
  for (int r = 0; r < 8; ++r) {
    int m = m0 + wm + r + 8 * hi;
    int n = n0 + wn + nl;
    float v00 = c00[r], v01 = c01[r], v10 = c10[r], v11 = c11[r];
    if (residual) {
      v00 += residual[(size_t)m * N + n];
      v01 += residual[(size_t)m * N + n + 16];
      v10 += residual[(size_t)(m + 16) * N + n];
      v11 += residual[(size_t)(m + 16) * N + n + 16];
    }
    C[(size_t)m * N + n]             = v00;
    C[(size_t)m * N + n + 16]        = v01;
    C[(size_t)(m + 16) * N + n]      = v10;
    C[(size_t)(m + 16) * N + n + 16] = v11;
  }
}

// ---------- RoPE + K/V concat (XL memory) ----------
__global__ void qkv_rope_kernel(const float* __restrict__ qkv,
                                const float* __restrict__ kxl,
                                const float* __restrict__ vxl,
                                const float* __restrict__ pos,
                                const float* __restrict__ fcos,
                                const float* __restrict__ fsin,
                                bf16_t* __restrict__ qb,
                                bf16_t* __restrict__ kc,
                                bf16_t* __restrict__ vc) {
  int idx = blockIdx.x * blockDim.x + threadIdx.x;
  const int total = B_ * S_ * H_ * (HD_ / 2);
  if (idx >= total) return;
  int j = idx & 31;            // pair index within head
  int h = (idx >> 5) & (H_ - 1);
  int s = (idx >> 9) & (S_ - 1);
  int b = idx >> 20;
  size_t catBase = (((size_t)(b * H_ + h)) * S_ + s) * HD_;
  if (s < T_) {
    size_t src  = ((size_t)b * T_ + s) * D_ + h * HD_ + 2 * j;
    size_t psrc = (size_t)s * D_ + h * HD_ + 2 * j;
    kc[catBase + 2 * j]     = f2bf(kxl[src]     + pos[psrc]);
    kc[catBase + 2 * j + 1] = f2bf(kxl[src + 1] + pos[psrc + 1]);
    vc[catBase + 2 * j]     = f2bf(vxl[src]);
    vc[catBase + 2 * j + 1] = f2bf(vxl[src + 1]);
  } else {
    int t = s - T_;
    size_t row = (size_t)b * T_ + t;
    size_t qo = row * (3 * D_) + h * HD_ + 2 * j;
    float c  = fcos[(size_t)t * (HD_ / 2) + j];
    float sn = fsin[(size_t)t * (HD_ / 2) + j];
    float qr = qkv[qo],          qi = qkv[qo + 1];
    float kr = qkv[qo + D_],     ki = qkv[qo + D_ + 1];
    float vr = qkv[qo + 2 * D_], vi = qkv[qo + 2 * D_ + 1];
    size_t qBase = (row * H_ + h) * HD_;
    qb[qBase + 2 * j]     = f2bf(qr * c - qi * sn);
    qb[qBase + 2 * j + 1] = f2bf(qr * sn + qi * c);
    kc[catBase + 2 * j]     = f2bf(kr * c - ki * sn);
    kc[catBase + 2 * j + 1] = f2bf(kr * sn + ki * c);
    vc[catBase + 2 * j]     = f2bf(vr);
    vc[catBase + 2 * j + 1] = f2bf(vi);
  }
}

// ---------- flash attention ----------
// grid: (T/128, B*H). 8 waves/block; each wave owns 16 query rows, online softmax.
__global__ void flash_attn_kernel(const bf16_t* __restrict__ qb,
                                  const bf16_t* __restrict__ kc,
                                  const bf16_t* __restrict__ vc,
                                  const int* __restrict__ causal_p,
                                  bf16_t* __restrict__ yb) {
  const int bh = blockIdx.y;
  const int b  = bh / H_;
  const int h  = bh % H_;
  const int wave = threadIdx.x >> 5;
  const int lane = threadIdx.x & 31;
  const int q0 = blockIdx.x * 128 + wave * 16;
  const int causal = causal_p[0];
  __shared__ __align__(16) bf16_t sK[32][72];        // [s][d]
  __shared__ __align__(16) bf16_t sV[64][40];        // [d][s] (transposed)
  __shared__ __align__(16) bf16_t sP[8][16][40];     // per-wave P staging

  const bf16_t* qbase = qb + ((size_t)(b * T_ + q0) * H_ + h) * HD_;
  v16bf aq0 = load_a_frag(qbase, H_ * HD_);
  v16bf aq1 = load_a_frag(qbase + 32, H_ * HD_);

  v8f o0 = {}, o1 = {}, o2 = {}, o3 = {};
  float Mrow[8], Lrow[8];
#pragma unroll
  for (int r = 0; r < 8; ++r) { Mrow[r] = -1e30f; Lrow[r] = 0.f; }

  const int hi = lane >> 4, nl = lane & 15;
  const size_t kvBase = (size_t)bh * S_ * HD_;
  const int ks = threadIdx.x >> 3, kcnk = (threadIdx.x & 7) * 8;  // K staging

  for (int s0 = 0; s0 < S_; s0 += 32) {
    cp_async_b128(&sK[ks][kcnk], kc + kvBase + (size_t)(s0 + ks) * HD_ + kcnk);
    for (int i = threadIdx.x; i < 32 * 64; i += 256) {
      int d = i & 63, s = i >> 6;
      sV[d][s] = vc[kvBase + (size_t)(s0 + s) * HD_ + d];
    }
    cp_async_wait();
    __syncthreads();

    v8f sc0 = {}, sc1 = {};
    sc0 = wmma_bf16(aq0, load_b_frag(&sK[0][0],  72), sc0);
    sc0 = wmma_bf16(aq1, load_b_frag(&sK[0][32], 72), sc0);
    sc1 = wmma_bf16(aq0, load_b_frag(&sK[16][0],  72), sc1);
    sc1 = wmma_bf16(aq1, load_b_frag(&sK[16][32], 72), sc1);

#pragma unroll
    for (int r = 0; r < 8; ++r) {
      float v0 = sc0[r] * 0.125f;   // 1/sqrt(64)
      float v1 = sc1[r] * 0.125f;
      if (causal) {
        int qi = q0 + r + 8 * hi;
        if (s0 + nl      > qi) v0 = -1e30f;
        if (s0 + 16 + nl > qi) v1 = -1e30f;
      }
      float rm = fmaxf(v0, v1);
#pragma unroll
      for (int off = 8; off >= 1; off >>= 1)
        rm = fmaxf(rm, __shfl_xor(rm, off, 32));
      float nm   = fmaxf(Mrow[r], rm);
      float corr = __expf(Mrow[r] - nm);
      float p0 = __expf(v0 - nm);
      float p1 = __expf(v1 - nm);
      float rs = p0 + p1;
#pragma unroll
      for (int off = 8; off >= 1; off >>= 1)
        rs += __shfl_xor(rs, off, 32);
      Lrow[r] = Lrow[r] * corr + rs;
      Mrow[r] = nm;
      o0[r] *= corr; o1[r] *= corr; o2[r] *= corr; o3[r] *= corr;
      int m = r + 8 * hi;
      sP[wave][m][nl]      = f2bf(p0);
      sP[wave][m][16 + nl] = f2bf(p1);
    }
    __syncthreads();

    v16bf pa = load_a_frag(&sP[wave][0][0], 40);
    o0 = wmma_bf16(pa, load_b_frag(&sV[0][0],  40), o0);
    o1 = wmma_bf16(pa, load_b_frag(&sV[16][0], 40), o1);
    o2 = wmma_bf16(pa, load_b_frag(&sV[32][0], 40), o2);
    o3 = wmma_bf16(pa, load_b_frag(&sV[48][0], 40), o3);
    __syncthreads();
  }

#pragma unroll
  for (int r = 0; r < 8; ++r) {
    int t = q0 + r + 8 * hi;
    float inv = 1.f / Lrow[r];
    size_t base = ((size_t)(b * T_ + t) * H_ + h) * HD_;
    yb[base + 0  + nl] = f2bf(o0[r] * inv);
    yb[base + 16 + nl] = f2bf(o1[r] * inv);
    yb[base + 32 + nl] = f2bf(o2[r] * inv);
    yb[base + 48 + nl] = f2bf(o3[r] * inv);
  }
}

// ---------- router: per-token logits, top-2, gates ----------
__global__ void router_kernel(const float* __restrict__ xf,
                              const float* __restrict__ rw,
                              const float* __restrict__ rb,
                              int* __restrict__ tix, float* __restrict__ tg) {
  int token = blockIdx.x * 8 + (threadIdx.x >> 5);
  int lane = threadIdx.x & 31;
  if (token >= M_) return;
  const float* xr = xf + (size_t)token * D_;
  float acc[E_];
#pragma unroll
  for (int e = 0; e < E_; ++e) acc[e] = 0.f;
  for (int d = lane; d < D_; d += 32) {
    float xv = xr[d];
    const float* wr = rw + (size_t)d * E_;
#pragma unroll
    for (int e = 0; e < E_; ++e) acc[e] += xv * wr[e];
  }
#pragma unroll
  for (int e = 0; e < E_; ++e) {
#pragma unroll
    for (int off = 16; off >= 1; off >>= 1)
      acc[e] += __shfl_xor(acc[e], off, 32);
  }
  if (lane == 0) {
#pragma unroll
    for (int e = 0; e < E_; ++e) acc[e] += rb[e];
    int i0 = 0; float b0 = acc[0];
    for (int e = 1; e < E_; ++e) if (acc[e] > b0) { b0 = acc[e]; i0 = e; }
    int i1 = -1; float b1 = -3e38f;
    for (int e = 0; e < E_; ++e)
      if (e != i0 && acc[e] > b1) { b1 = acc[e]; i1 = e; }
    float e1 = __expf(b1 - b0);
    float g0 = 1.f / (1.f + e1);
    tix[token * 2]     = i0;  tg[token * 2]     = g0;
    tix[token * 2 + 1] = i1;  tg[token * 2 + 1] = e1 * g0;
  }
}

// Deterministic per-expert compaction: one warp per expert, ballot prefix sums.
// etok entry encodes token*2 + slot.
__global__ void build_lists_kernel(const int* __restrict__ tix,
                                   const float* __restrict__ tg,
                                   int* __restrict__ counts,
                                   int* __restrict__ etok,
                                   float* __restrict__ egate) {
  int e    = threadIdx.x >> 5;
  int lane = threadIdx.x & 31;
  int pos = 0;
  for (int base = 0; base < M_; base += 32) {
    int t = base + lane;
    int i0 = tix[t * 2], i1 = tix[t * 2 + 1];
    bool match = (i0 == e) || (i1 == e);
    unsigned int bal = (unsigned int)__ballot(match);
    int pre = __popc(bal & ((1u << lane) - 1u));
    if (match) {
      int slot = (i0 == e) ? 0 : 1;
      etok[e * M_ + pos + pre]  = t * 2 + slot;
      egate[e * M_ + pos + pre] = slot ? tg[t * 2 + 1] : tg[t * 2];
    }
    pos += __popc(bal);
  }
  if (lane == 0) counts[e] = pos;
}

// ---------- MoE grouped GEMM 1: H = silu(Xg @ w1[e]) * (Xg @ w3[e]) ----------
// Block tile 64(M) x 128(N); per wave 32x32 against BOTH w1 and w3 (8 WMMA/K-step).
__global__ void moe_gemm13_kernel(const bf16_t* __restrict__ X,
                                  const bf16_t* __restrict__ W1,
                                  const bf16_t* __restrict__ W3,
                                  const int* __restrict__ counts,
                                  const int* __restrict__ etok,
                                  bf16_t* __restrict__ Hbuf) {
  const int e = blockIdx.z;
  const int cnt = counts[e];
  const int m0 = blockIdx.y * 64;
  if (m0 >= cnt) return;                       // block-uniform
  const int n0 = blockIdx.x * 128;
  const int tid = threadIdx.x;
  __shared__ int stok[64];
  __shared__ __align__(16) bf16_t sA[64][40];
  __shared__ __align__(16) bf16_t sB1[128][40];
  __shared__ __align__(16) bf16_t sB3[128][40];
  if (tid < 64)
    stok[tid] = (m0 + tid < cnt) ? (etok[e * M_ + m0 + tid] >> 1) : 0;
  __syncthreads();
  const int wave = tid >> 5, lane = tid & 31;
  const int wm = (wave >> 2) * 32, wn = (wave & 3) * 32;
  const int am = tid >> 2, ac = (tid & 3) * 8;
  const size_t wOff = (size_t)e * D_ * FH_;
  v8f c1[2][2], c3[2][2];
#pragma unroll
  for (int mi = 0; mi < 2; ++mi)
#pragma unroll
    for (int ni = 0; ni < 2; ++ni) { c1[mi][ni] = {}; c3[mi][ni] = {}; }
  for (int k0 = 0; k0 < D_; k0 += 32) {
    cp_async_b128(&sA[am][ac], X + (size_t)stok[am] * D_ + k0 + ac);
    for (int i = tid; i < 128 * 32; i += 256) {
      int n = i & 127, k = i >> 7;
      size_t gi = wOff + (size_t)(k0 + k) * FH_ + n0 + n;
      sB1[n][k] = W1[gi];
      sB3[n][k] = W3[gi];
    }
    cp_async_wait();
    __syncthreads();
    v16bf a0  = load_a_frag(&sA[wm][0], 40);
    v16bf a1  = load_a_frag(&sA[wm + 16][0], 40);
    v16bf b10 = load_b_frag(&sB1[wn][0], 40);
    v16bf b11 = load_b_frag(&sB1[wn + 16][0], 40);
    v16bf b30 = load_b_frag(&sB3[wn][0], 40);
    v16bf b31 = load_b_frag(&sB3[wn + 16][0], 40);
    c1[0][0] = wmma_bf16(a0, b10, c1[0][0]);
    c1[0][1] = wmma_bf16(a0, b11, c1[0][1]);
    c1[1][0] = wmma_bf16(a1, b10, c1[1][0]);
    c1[1][1] = wmma_bf16(a1, b11, c1[1][1]);
    c3[0][0] = wmma_bf16(a0, b30, c3[0][0]);
    c3[0][1] = wmma_bf16(a0, b31, c3[0][1]);
    c3[1][0] = wmma_bf16(a1, b30, c3[1][0]);
    c3[1][1] = wmma_bf16(a1, b31, c3[1][1]);
    __syncthreads();
  }
  const int hi = lane >> 4, nl = lane & 15;
#pragma unroll
  for (int mi = 0; mi < 2; ++mi) {
#pragma unroll
    for (int r = 0; r < 8; ++r) {
      int m = wm + mi * 16 + r + 8 * hi;
      if (m0 + m < cnt) {
#pragma unroll
        for (int ni = 0; ni < 2; ++ni) {
          float g1 = c1[mi][ni][r], g3 = c3[mi][ni][r];
          float hv = g1 / (1.f + __expf(-g1)) * g3;  // silu(g1)*g3
          Hbuf[((size_t)e * M_ + m0 + m) * FH_ + n0 + wn + ni * 16 + nl] = f2bf(hv);
        }
      }
    }
  }
}

// ---------- MoE grouped GEMM 2: Y = H @ w2[e], scattered by gate into per-slot buffers
__global__ void moe_gemm2_kernel(const bf16_t* __restrict__ Hbuf,
                                 const bf16_t* __restrict__ W2,
                                 const int* __restrict__ counts,
                                 const int* __restrict__ etok,
                                 const float* __restrict__ egate,
                                 float* __restrict__ moeT) {  // [2][M_][D_]
  const int e = blockIdx.z;
  const int cnt = counts[e];
  const int m0 = blockIdx.y * 64;
  if (m0 >= cnt) return;
  const int n0 = blockIdx.x * 128;
  const int tid = threadIdx.x;
  __shared__ __align__(16) bf16_t sA[64][40];
  __shared__ __align__(16) bf16_t sB[128][40];
  const int wave = tid >> 5, lane = tid & 31;
  const int wm = (wave >> 2) * 32, wn = (wave & 3) * 32;
  const int am = tid >> 2, ac = (tid & 3) * 8;
  const size_t hOff = (size_t)e * M_ * FH_;
  const size_t wOff = (size_t)e * FH_ * D_;
  v8f c00 = {}, c01 = {}, c10 = {}, c11 = {};
  for (int k0 = 0; k0 < FH_; k0 += 32) {
    // rows >= cnt read harmless in-bounds garbage; their outputs are discarded
    cp_async_b128(&sA[am][ac], Hbuf + hOff + (size_t)(m0 + am) * FH_ + k0 + ac);
    for (int i = tid; i < 128 * 32; i += 256) {
      int n = i & 127, k = i >> 7;
      sB[n][k] = W2[wOff + (size_t)(k0 + k) * D_ + n0 + n];
    }
    cp_async_wait();
    __syncthreads();
    v16bf a0 = load_a_frag(&sA[wm][0], 40);
    v16bf a1 = load_a_frag(&sA[wm + 16][0], 40);
    v16bf b0 = load_b_frag(&sB[wn][0], 40);
    v16bf b1 = load_b_frag(&sB[wn + 16][0], 40);
    c00 = wmma_bf16(a0, b0, c00);
    c01 = wmma_bf16(a0, b1, c01);
    c10 = wmma_bf16(a1, b0, c10);
    c11 = wmma_bf16(a1, b1, c11);
    __syncthreads();
  }
  const int hi = lane >> 4, nl = lane & 15;
#pragma unroll
  for (int mi = 0; mi < 2; ++mi) {
#pragma unroll
    for (int r = 0; r < 8; ++r) {
      int m = m0 + wm + mi * 16 + r + 8 * hi;
      if (m < cnt) {
        int entry = etok[e * M_ + m];
        int tok = entry >> 1, slot = entry & 1;
        float g = egate[e * M_ + m];
#pragma unroll
        for (int ni = 0; ni < 2; ++ni) {
          float v = mi ? (ni ? c11[r] : c10[r]) : (ni ? c01[r] : c00[r]);
          moeT[(size_t)slot * M_ * D_ + (size_t)tok * D_ + n0 + wn + ni * 16 + nl] =
              g * v;
        }
      }
    }
  }
}

// ---------- host orchestration ----------
extern "C" void kernel_launch(void* const* d_in, const int* in_sizes, int n_in,
                              void* d_out, int out_size, void* d_ws, size_t ws_size,
                              hipStream_t stream) {
  (void)in_sizes; (void)n_in; (void)out_size; (void)ws_size;
  const float* q     = (const float*)d_in[0];
  const float* fcos  = (const float*)d_in[3];
  const float* fsin  = (const float*)d_in[4];
  const float* kxl   = (const float*)d_in[5];
  const float* vxl   = (const float*)d_in[6];
  const float* Wqkv  = (const float*)d_in[7];
  const float* Wproj = (const float*)d_in[8];
  const float* pos   = (const float*)d_in[9];
  const float* anw   = (const float*)d_in[10];
  const float* fnw   = (const float*)d_in[11];
  const float* rw    = (const float*)d_in[12];
  const float* rb    = (const float*)d_in[13];
  const float* w1    = (const float*)d_in[14];
  const float* w2    = (const float*)d_in[15];
  const float* w3    = (const float*)d_in[16];
  const int*   causal = (const int*)d_in[17];
  float* out = (float*)d_out;

  char* ws = (char*)d_ws;
  size_t off = 0;
  auto alloc = [&](size_t bytes) -> void* {
    void* p = ws + off;
    off += (bytes + 255) & ~(size_t)255;
    return p;
  };

  bf16_t* WqkvB = (bf16_t*)alloc((size_t)D_ * 3 * D_ * 2);
  bf16_t* WprojB= (bf16_t*)alloc((size_t)D_ * D_ * 2);
  bf16_t* w1B   = (bf16_t*)alloc((size_t)E_ * D_ * FH_ * 2);
  bf16_t* w2B   = (bf16_t*)alloc((size_t)E_ * FH_ * D_ * 2);
  bf16_t* w3B   = (bf16_t*)alloc((size_t)E_ * D_ * FH_ * 2);
  float*  xnF   = (float*)alloc((size_t)M_ * D_ * 4);
  bf16_t* xnB   = (bf16_t*)alloc((size_t)M_ * D_ * 2);
  float*  qkvF  = (float*)alloc((size_t)M_ * 3 * D_ * 4);
  bf16_t* qB    = (bf16_t*)alloc((size_t)M_ * D_ * 2);
  bf16_t* Kc    = (bf16_t*)alloc((size_t)B_ * H_ * S_ * HD_ * 2);
  bf16_t* Vc    = (bf16_t*)alloc((size_t)B_ * H_ * S_ * HD_ * 2);
  bf16_t* yB    = (bf16_t*)alloc((size_t)M_ * D_ * 2);
  float*  hF    = (float*)alloc((size_t)M_ * D_ * 4);
  float*  xfF   = (float*)alloc((size_t)M_ * D_ * 4);
  bf16_t* xfB   = (bf16_t*)alloc((size_t)M_ * D_ * 2);
  int*    tix   = (int*)alloc((size_t)M_ * 2 * 4);
  float*  tg    = (float*)alloc((size_t)M_ * 2 * 4);
  int*    counts= (int*)alloc(E_ * 4);
  int*    etok  = (int*)alloc((size_t)E_ * M_ * 4);
  float*  egate = (float*)alloc((size_t)E_ * M_ * 4);
  float*  moeT  = (float*)alloc((size_t)2 * M_ * D_ * 4);
  bf16_t* Hbuf  = (bf16_t*)alloc((size_t)E_ * M_ * FH_ * 2);

  auto cvt = [&](const float* src, bf16_t* dst, size_t n) {
    int blocks = (int)((n + 2047) / 2048);
    cvt_f32_bf16_kernel<<<blocks, 256, 0, stream>>>(src, dst, (int)n);
  };
  cvt(Wqkv,  WqkvB,  (size_t)D_ * 3 * D_);
  cvt(Wproj, WprojB, (size_t)D_ * D_);
  cvt(w1, w1B, (size_t)E_ * D_ * FH_);
  cvt(w2, w2B, (size_t)E_ * FH_ * D_);
  cvt(w3, w3B, (size_t)E_ * D_ * FH_);

  // 1) xn = rmsnorm(q, attn_norm_w)
  rmsnorm_kernel<<<M_, 256, 0, stream>>>(q, anw, xnF, xnB);
  // 2) qkv = xn @ W_qkv
  gemm_bf16_kernel<<<dim3(3 * D_ / 128, M_ / 64), 256, 0, stream>>>(
      xnB, WqkvB, nullptr, qkvF, M_, 3 * D_, D_);
  // 3) RoPE + XL concat
  qkv_rope_kernel<<<(B_ * S_ * H_ * 32) / 256, 256, 0, stream>>>(
      qkvF, kxl, vxl, pos, fcos, fsin, qB, Kc, Vc);
  // 4) flash attention -> y
  flash_attn_kernel<<<dim3(T_ / 128, B_ * H_), 256, 0, stream>>>(
      qB, Kc, Vc, causal, yB);
  // 5) h = xn + y @ W_proj
  gemm_bf16_kernel<<<dim3(D_ / 128, M_ / 64), 256, 0, stream>>>(
      yB, WprojB, xnF, hF, M_, D_, D_);
  // 6) xf = rmsnorm(h, ffn_norm_w)
  rmsnorm_kernel<<<M_, 256, 0, stream>>>(hF, fnw, xfF, xfB);
  // 7) router + deterministic expert compaction
  router_kernel<<<M_ / 8, 256, 0, stream>>>(xfF, rw, rb, tix, tg);
  build_lists_kernel<<<1, 256, 0, stream>>>(tix, tg, counts, etok, egate);
  // 8) MoE grouped GEMMs
  moe_gemm13_kernel<<<dim3(FH_ / 128, M_ / 64, E_), 256, 0, stream>>>(
      xfB, w1B, w3B, counts, etok, Hbuf);
  moe_gemm2_kernel<<<dim3(D_ / 128, M_ / 64, E_), 256, 0, stream>>>(
      Hbuf, w2B, counts, etok, egate, moeT);
  // 9) out = h + moe_slot0 + moe_slot1
  final_add_kernel<<<(M_ * D_) / 2048, 256, 0, stream>>>(
      hF, moeT, moeT + (size_t)M_ * D_, out, M_ * D_);
}
